// SelfAttentionBlock_44143673868620
// MI455X (gfx1250) — matrix operand order repaired and checked by reading, not compiled, not run on hardware
//
#include <hip/hip_runtime.h>

// ---------- types ----------
typedef __attribute__((ext_vector_type(16))) __bf16 v16bf;
typedef __attribute__((ext_vector_type(8)))  float  v8f;
typedef unsigned int u32x4 __attribute__((ext_vector_type(4)));

union Frag {          // 16 bf16 halves = one WMMA A/B fragment (per lane)
    v16bf v;
    u32x4 q[2];
};

// native fptrunc -> hardware bf16 convert (gfx1250 has native bf16 datapath)
__device__ __forceinline__ unsigned short f2bf(float f) {
    union { __bf16 h; unsigned short u; } cv;
    cv.h = (__bf16)f;
    return cv.u;
}

#define WMMA_BF16(A, B, C) \
    __builtin_amdgcn_wmma_f32_16x16x32_bf16(false, (A), false, (B), (short)0, (C), false, false)

// ---------- constants ----------
#define NB     2
#define LSEQ   2304        // 48*48
#define STOK   4608        // NB*LSEQ
#define DM     1024
#define CF     768
#define NH     16
#define DH     64

// ================= kernel 1: ada = cond @ norm_w^T + 1 =================
__global__ void ada_kernel(const float* __restrict__ cond,
                           const float* __restrict__ norm_w,
                           float* __restrict__ ada) {
    int j = blockIdx.x * blockDim.x + threadIdx.x;   // 0..2047
    int b = j >> 10, col = j & 1023;
    const float* c = cond + b * CF;
    const float* wr = norm_w + (size_t)col * CF;
    float acc = 0.f;
    for (int i = 0; i < CF; ++i) acc += c[i] * wr[i];
    ada[j] = acc + 1.0f;
}

// ============ kernel 2: RMS-norm * ada -> bf16 xn (one block/token) ============
__global__ void rmsnorm_kernel(const float* __restrict__ x,
                               const float* __restrict__ ada,
                               unsigned short* __restrict__ xn) {
    int t = blockIdx.x;                 // 0..4607
    int b = t / LSEQ;
    const float* row = x + (size_t)t * DM;
    int i0 = threadIdx.x * 4;
    float4 v = *(const float4*)(row + i0);
    float s = v.x * v.x + v.y * v.y + v.z * v.z + v.w * v.w;
    for (int off = 16; off > 0; off >>= 1) s += __shfl_xor(s, off);
    __shared__ float red[8];
    if ((threadIdx.x & 31) == 0) red[threadIdx.x >> 5] = s;
    __syncthreads();
    float total = 0.f;
    for (int k = 0; k < 8; ++k) total += red[k];
    float inv = rsqrtf(total * (1.0f / DM) + 1e-6f);
    const float* arow = ada + (size_t)b * DM;
    union { unsigned short h[4]; unsigned long long u; } pk;
    pk.h[0] = f2bf(v.x * arow[i0 + 0] * inv);
    pk.h[1] = f2bf(v.y * arow[i0 + 1] * inv);
    pk.h[2] = f2bf(v.z * arow[i0 + 2] * inv);
    pk.h[3] = f2bf(v.w * arow[i0 + 3] * inv);
    *(unsigned long long*)(xn + (size_t)t * DM + i0) = pk.u;
}

// ============ kernel 3: f32 -> bf16 weight cast ============
__global__ void cast_bf16_kernel(const float* __restrict__ in,
                                 unsigned short* __restrict__ out, int n) {
    int i = blockIdx.x * blockDim.x + threadIdx.x;
    if (i < n) out[i] = f2bf(in[i]);
}

// ============ kernel 4/7: WMMA GEMM  C[M,N] = A[M,K] * B[N,K]^T (+resid) ============
// A,B bf16 row-major (rows of length K). BM=BN=128, BK=32, 256 thr = 8 waves,
// wave computes 64x32 via 4x2 tiles of v_wmma_f32_16x16x32_bf16.
template <int EPI>
__global__ void __launch_bounds__(256)
gemm_wmma_kernel(const unsigned short* __restrict__ A,
                 const unsigned short* __restrict__ B,
                 float* __restrict__ C,
                 const float* __restrict__ resid,
                 int M, int N, int K) {
    __shared__ __attribute__((aligned(16))) unsigned short As[128 * 32];
    __shared__ __attribute__((aligned(16))) unsigned short Bs[128 * 32];

    int nblk = N >> 7;
    int bm = blockIdx.x / nblk;
    int bn = blockIdx.x % nblk;
    int tid = threadIdx.x, wid = tid >> 5, lane = tid & 31;
    int wm = wid >> 2, wn = wid & 3;          // 2x4 wave grid
    int l16 = lane & 15, g = lane >> 4;

    v8f acc[4][2];
#pragma unroll
    for (int i = 0; i < 4; ++i)
#pragma unroll
        for (int j = 0; j < 2; ++j)
            acc[i][j] = (v8f){0.f, 0.f, 0.f, 0.f, 0.f, 0.f, 0.f, 0.f};

    int lrow = tid >> 1;                      // 0..127
    int lcol = (tid & 1) * 16;                // 0 / 16
    const unsigned short* Ag = A + (size_t)(bm * 128 + lrow) * K + lcol;
    const unsigned short* Bg = B + (size_t)(bn * 128 + lrow) * K + lcol;
    unsigned short* AsW = As + lrow * 32 + lcol;
    unsigned short* BsW = Bs + lrow * 32 + lcol;

    for (int k0 = 0; k0 < K; k0 += 32) {
        __syncthreads();
        *(u32x4*)(AsW)     = *(const u32x4*)(Ag + k0);
        *(u32x4*)(AsW + 8) = *(const u32x4*)(Ag + k0 + 8);
        *(u32x4*)(BsW)     = *(const u32x4*)(Bg + k0);
        *(u32x4*)(BsW + 8) = *(const u32x4*)(Bg + k0 + 8);
        if (k0 + 32 < K) {                    // prefetch next K-slab (global_prefetch_b8)
            __builtin_prefetch(Ag + k0 + 32, 0, 1);
            __builtin_prefetch(Bg + k0 + 32, 0, 1);
        }
        __syncthreads();

        Frag bfg[2];
#pragma unroll
        for (int j = 0; j < 2; ++j) {
            int r = wn * 32 + j * 16 + l16;
            bfg[j].q[0] = *(const u32x4*)(Bs + r * 32 + g * 8);
            bfg[j].q[1] = *(const u32x4*)(Bs + r * 32 + 16 + g * 8);
        }
#pragma unroll
        for (int i = 0; i < 4; ++i) {         // one A-frag live at a time
            Frag af;
            int r = wm * 64 + i * 16 + l16;
            af.q[0] = *(const u32x4*)(As + r * 32 + g * 8);
            af.q[1] = *(const u32x4*)(As + r * 32 + 16 + g * 8);
            acc[i][0] = WMMA_BF16(af.v, bfg[0].v, acc[i][0]);
            acc[i][1] = WMMA_BF16(af.v, bfg[1].v, acc[i][1]);
        }
    }

#pragma unroll
    for (int i = 0; i < 4; ++i)
#pragma unroll
        for (int j = 0; j < 2; ++j) {
            int row0 = bm * 128 + wm * 64 + i * 16 + g * 8;
            int col  = bn * 128 + wn * 32 + j * 16 + l16;
#pragma unroll
            for (int r = 0; r < 8; ++r) {
                size_t idx = (size_t)(row0 + r) * N + col;
                float val = acc[i][j][r];
                if (EPI == 1) val += resid[idx];
                C[idx] = val;
            }
        }
}

// ============ kernel 5: q/k norm + RoPE + head-major scatter ============
__device__ __forceinline__ float rope_apply(float p0, float p1, int j, float x, float partner) {
    int idx = j & 15;
    float fr = 3.14159265358979f * __expf((float)(idx & 7) * 0.28782313662425572f); // pi*10^(i/8)
    float f  = ((idx < 8) ? p0 : p1) * fr;
    float rot = (j < 16) ? -partner : partner;
    return x * __cosf(f) + rot * __sinf(f);
}

__global__ void qk_rope_kernel(const float* __restrict__ qkv,
                               const float* __restrict__ pos,
                               const float* __restrict__ scale,
                               unsigned short* __restrict__ qh,
                               unsigned short* __restrict__ kh,
                               unsigned short* __restrict__ vh) {
    int gw = blockIdx.x * 8 + (threadIdx.x >> 5);  // one wave per (b,t,head)
    int lane = threadIdx.x & 31;
    int hh = gw & 15;
    int t  = (gw >> 4) % LSEQ;
    int b  = gw / (NH * LSEQ);
    const float* base = qkv + (size_t)(b * LSEQ + t) * (3 * DM) + hh * DH;
    int d0 = lane * 2;

    float q0 = base[d0],          q1 = base[d0 + 1];
    float k0 = base[DM + d0],     k1 = base[DM + d0 + 1];
    float v0 = base[2 * DM + d0], v1 = base[2 * DM + d0 + 1];

    float sq = q0 * q0 + q1 * q1;
    float sk = k0 * k0 + k1 * k1;
    for (int off = 1; off < 32; off <<= 1) { sq += __shfl_xor(sq, off); sk += __shfl_xor(sk, off); }
    float ss = sqrtf(scale[hh]);
    float rq = ss * rsqrtf(sq + 1e-6f), rk = ss * rsqrtf(sk + 1e-6f);
    q0 *= rq; q1 *= rq; k0 *= rk; k1 *= rk;

    float p0 = pos[t * 2], p1 = pos[t * 2 + 1];
    float q0p = __shfl_xor(q0, 8), q1p = __shfl_xor(q1, 8);
    float k0p = __shfl_xor(k0, 8), k1p = __shfl_xor(k1, 8);
    if (lane < 16) {   // dims 0..31 get rotary, dims 32..63 pass through
        float nq0 = rope_apply(p0, p1, d0,     q0, q0p);
        float nq1 = rope_apply(p0, p1, d0 + 1, q1, q1p);
        float nk0 = rope_apply(p0, p1, d0,     k0, k0p);
        float nk1 = rope_apply(p0, p1, d0 + 1, k1, k1p);
        q0 = nq0; q1 = nq1; k0 = nk0; k1 = nk1;
    }

    size_t ob = ((size_t)((b * NH + hh) * LSEQ + t)) * DH + d0;
    qh[ob] = f2bf(q0); qh[ob + 1] = f2bf(q1);
    kh[ob] = f2bf(k0); kh[ob + 1] = f2bf(k1);
    vh[ob] = f2bf(v0); vh[ob + 1] = f2bf(v1);
}

// ============ kernel 6: flash attention (WMMA, constant-shift softmax) ============
// |q|=|k|=sqrt(scale[h])  =>  |q.k| <= scale[h]; softmax is shift-invariant, so a
// constant shift of scale[h] is exact and removes the online-max machinery entirely.
// Row sums are accumulated with a WMMA against a constant all-ones B fragment.
// grid = 32 (b*head) * 18 q-tiles; block = 256 = 8 waves; wave owns 16 q rows.
__global__ void __launch_bounds__(256)
flash_attn_kernel(const unsigned short* __restrict__ qh,
                  const unsigned short* __restrict__ kh,
                  const unsigned short* __restrict__ vh,
                  const float* __restrict__ scale,
                  unsigned short* __restrict__ O) {
    __shared__ __attribute__((aligned(16))) unsigned short Ks[32 * 64];   // [key][dim]
    __shared__ __attribute__((aligned(16))) unsigned short Vt[64 * 32];   // [dim][key]
    __shared__ __attribute__((aligned(16))) unsigned short Ps[8][16 * 32];// per-wave P

    int bh = blockIdx.x / 18;          // 0..31
    int qt = blockIdx.x % 18;
    int b = bh >> 4, hh = bh & 15;
    int tid = threadIdx.x, wid = tid >> 5, lane = tid & 31;
    int l16 = lane & 15, g = lane >> 4;

    const unsigned short* Q  = qh + (size_t)bh * LSEQ * DH;
    const unsigned short* Kg = kh + (size_t)bh * LSEQ * DH;
    const unsigned short* Vg = vh + (size_t)bh * LSEQ * DH;
    float shift = scale[hh];           // upper bound on every score

    int qrow = qt * 128 + wid * 16 + l16;
    Frag aq[2];
#pragma unroll
    for (int c = 0; c < 2; ++c) {
        aq[c].q[0] = *(const u32x4*)(Q + (size_t)qrow * DH + c * 32 + g * 8);
        aq[c].q[1] = *(const u32x4*)(Q + (size_t)qrow * DH + c * 32 + 16 + g * 8);
    }

    Frag ones;                         // bf16 1.0 broadcast: 32x16 all-ones B matrix
    ones.q[0] = (u32x4){0x3F803F80u, 0x3F803F80u, 0x3F803F80u, 0x3F803F80u};
    ones.q[1] = ones.q[0];

    v8f ot[4];
#pragma unroll
    for (int d = 0; d < 4; ++d) ot[d] = (v8f){0.f, 0.f, 0.f, 0.f, 0.f, 0.f, 0.f, 0.f};
    v8f lacc = (v8f){0.f, 0.f, 0.f, 0.f, 0.f, 0.f, 0.f, 0.f};

    int kkey = tid & 31;               // co-op load mapping
    int dbase = (tid >> 5) * 8;

    for (int c0 = 0; c0 < LSEQ; c0 += 32) {
        __syncthreads();
        // K chunk row-major; V chunk transposed into Vt
        *(u32x4*)(Ks + kkey * 64 + dbase) = *(const u32x4*)(Kg + (size_t)(c0 + kkey) * DH + dbase);
        union { u32x4 u; unsigned short h[8]; } vv;
        vv.u = *(const u32x4*)(Vg + (size_t)(c0 + kkey) * DH + dbase);
#pragma unroll
        for (int e = 0; e < 8; ++e) Vt[(dbase + e) * 32 + kkey] = vv.h[e];
        if (c0 + 32 < LSEQ) {          // prefetch next chunk (global_prefetch_b8)
            __builtin_prefetch(Kg + (size_t)(c0 + 32 + kkey) * DH + dbase, 0, 1);
            __builtin_prefetch(Vg + (size_t)(c0 + 32 + kkey) * DH + dbase, 0, 1);
        }
        __syncthreads();

        // scores: 16 q x 32 k  (two 16x16 tiles, head-dim 64 = 2 WMMA each),
        // then p = exp(s - shift) straight into the wave-private P staging tile.
        unsigned short* P = Ps[wid];
#pragma unroll
        for (int st = 0; st < 2; ++st) {
            Frag bk0, bk1;
            int kr = st * 16 + l16;
            bk0.q[0] = *(const u32x4*)(Ks + kr * 64 + g * 8);
            bk0.q[1] = *(const u32x4*)(Ks + kr * 64 + 16 + g * 8);
            bk1.q[0] = *(const u32x4*)(Ks + kr * 64 + 32 + g * 8);
            bk1.q[1] = *(const u32x4*)(Ks + kr * 64 + 32 + 16 + g * 8);
            v8f c = (v8f){0.f, 0.f, 0.f, 0.f, 0.f, 0.f, 0.f, 0.f};
            c = WMMA_BF16(aq[0].v, bk0.v, c);
            c = WMMA_BF16(aq[1].v, bk1.v, c);
#pragma unroll
            for (int r = 0; r < 8; ++r)
                P[(r + g * 8) * 32 + st * 16 + l16] = f2bf(__expf(c[r] - shift));
        }

        // reload P in A-fragment layout (C-layout -> A-layout transpose via LDS)
        Frag ap;
        ap.q[0] = *(const u32x4*)(P + l16 * 32 + g * 8);
        ap.q[1] = *(const u32x4*)(P + l16 * 32 + 16 + g * 8);

        // row sums via ones-matrix WMMA (replaces 32 ds_bpermute per chunk)
        lacc = WMMA_BF16(ap.v, ones.v, lacc);

        // o += P(16x32) * V(32x64)
#pragma unroll
        for (int d = 0; d < 4; ++d) {
            Frag bv;
            int vr = d * 16 + l16;
            bv.q[0] = *(const u32x4*)(Vt + vr * 32 + g * 8);
            bv.q[1] = *(const u32x4*)(Vt + vr * 32 + 16 + g * 8);
            ot[d] = WMMA_BF16(ap.v, bv.v, ot[d]);
        }
    }

    // normalize and store O[token][head*64 + dim] as bf16
    float invl[8];
#pragma unroll
    for (int r = 0; r < 8; ++r) invl[r] = 1.0f / lacc[r];
#pragma unroll
    for (int d = 0; d < 4; ++d)
#pragma unroll
        for (int r = 0; r < 8; ++r) {
            int m = r + 8 * g;
            int token = b * LSEQ + qt * 128 + wid * 16 + m;
            O[(size_t)token * DM + hh * DH + d * 16 + l16] = f2bf(ot[d][r] * invl[r]);
        }
}

// ================= launch =================
extern "C" void kernel_launch(void* const* d_in, const int* in_sizes, int n_in,
                              void* d_out, int out_size, void* d_ws, size_t ws_size,
                              hipStream_t stream) {
    const float* x      = (const float*)d_in[0];
    const float* pos    = (const float*)d_in[1];
    const float* cond   = (const float*)d_in[2];
    const float* norm_w = (const float*)d_in[3];
    const float* qkv_w  = (const float*)d_in[4];
    const float* scale  = (const float*)d_in[5];
    const float* out_w  = (const float*)d_in[6];

    char* w = (char*)d_ws;
    auto alloc = [&](size_t bytes) -> void* {
        void* p = (void*)w;
        w += (bytes + 255) & ~(size_t)255;
        return p;
    };
    float*          ada  = (float*)alloc((size_t)NB * DM * 4);
    unsigned short* xn   = (unsigned short*)alloc((size_t)STOK * DM * 2);
    unsigned short* qkvw = (unsigned short*)alloc((size_t)3 * DM * DM * 2);
    unsigned short* outw = (unsigned short*)alloc((size_t)DM * DM * 2);
    float*          qkv  = (float*)alloc((size_t)STOK * 3 * DM * 4);
    unsigned short* qhb  = (unsigned short*)alloc((size_t)STOK * DM * 2);
    unsigned short* khb  = (unsigned short*)alloc((size_t)STOK * DM * 2);
    unsigned short* vhb  = (unsigned short*)alloc((size_t)STOK * DM * 2);
    unsigned short* Obuf = (unsigned short*)alloc((size_t)STOK * DM * 2);

    ada_kernel<<<(NB * DM) / 256, 256, 0, stream>>>(cond, norm_w, ada);
    rmsnorm_kernel<<<STOK, 256, 0, stream>>>(x, ada, xn);
    cast_bf16_kernel<<<(3 * DM * DM + 255) / 256, 256, 0, stream>>>(qkv_w, qkvw, 3 * DM * DM);
    cast_bf16_kernel<<<(DM * DM + 255) / 256, 256, 0, stream>>>(out_w, outw, DM * DM);

    // qkv = xn @ qkv_w^T : M=4608, N=3072, K=1024
    gemm_wmma_kernel<0><<<(STOK / 128) * (3 * DM / 128), 256, 0, stream>>>(
        xn, qkvw, qkv, nullptr, STOK, 3 * DM, DM);

    qk_rope_kernel<<<(NB * LSEQ * NH) / 8, 256, 0, stream>>>(qkv, pos, scale, qhb, khb, vhb);

    flash_attn_kernel<<<NB * NH * (LSEQ / 128), 256, 0, stream>>>(qhb, khb, vhb, scale, Obuf);

    // out = O @ out_w^T + skip : M=4608, N=1024, K=1024
    gemm_wmma_kernel<1><<<(STOK / 128) * (DM / 128), 256, 0, stream>>>(
        Obuf, outw, (float*)d_out, x, STOK, DM, DM);
}
